// VectorQuantizer_18794776888090
// MI455X (gfx1250) — matrix operand order repaired and checked by reading, not compiled, not run on hardware
//
#include <hip/hip_runtime.h>

// ---------------------------------------------------------------------------
// VQ-VAE eval forward for MI455X (gfx1250, wave32, WMMA + async-LDS copies).
// Distance GEMM (65536 x 2048 x 128) via v_wmma_f32_16x16x32_bf16 fused with
// per-row argmin; B tiles staged in LDS (double-buffered) with
// global_load_async_to_lds_b128 (inline asm, ASYNCcnt-tracked) so one L2
// fetch serves all 8 waves. ||x||^2 is row-constant and dropped from the
// argmin; ||e||^2 bias and the commitment loss stay in f32.
// ---------------------------------------------------------------------------

typedef __attribute__((ext_vector_type(16))) __bf16 v16bf;
typedef __attribute__((ext_vector_type(8)))  __bf16 v8bf;
typedef __attribute__((ext_vector_type(8)))  float  v8f;

typedef __attribute__((address_space(3))) void lvoid;

#define BB 64
#define DD 128
#define TT 1024
#define KK 2048
#define NN (BB * TT)          // 65536 rows
#define NTILE 128             // code tiles of 16 codes
#define ROWS 128              // rows per block

// workspace byte offsets
#define WS_CBF    0           // K*D bf16 fragment-packed codebook: 524288 B
#define WS_NORMS  524288      // K f32 code norms: 8192 B
#define WS_COUNTS 532480      // K f32 histogram: 8192 B
#define WS_LOSS   540672      // 1 f32 loss accumulator
#define WS_IDX    540736      // N int32 argmin indices: 262144 B

// --------------------------- async-LDS helpers -----------------------------
// Inline asm (portable across ROCm 7.2 / amdgpu-toolchain): the clang builtin
// exists but its params are in the unspellable cuda_device/cuda_shared LangAS.
__device__ __forceinline__ void async_ld_b128(const void* g, void* l) {
  unsigned loff = (unsigned)(size_t)(lvoid*)l;   // LDS byte offset (AS3 ptrtoint)
  asm volatile("global_load_async_to_lds_b128 %0, %1, off"
               :: "v"(loff), "v"(g) : "memory");
}
__device__ __forceinline__ void wait_async_0() {
  asm volatile("s_wait_asynccnt 0x0" ::: "memory");
}
__device__ __forceinline__ void wait_async_le1() {
  asm volatile("s_wait_asynccnt 0x1" ::: "memory");
}

// ---------------------------------------------------------------------------
// K1: code norms ||e_k||^2 (blocks 0..7) + zero counts/loss (blocks 8..15)
// ---------------------------------------------------------------------------
__global__ __launch_bounds__(256) void k_prep(const float* __restrict__ cb,
                                              float* __restrict__ norms,
                                              float* __restrict__ counts,
                                              float* __restrict__ loss) {
  int g = blockIdx.x * 256 + threadIdx.x;
  if (blockIdx.x < 8) {                       // g = code 0..2047
    const float* r = cb + (size_t)g * DD;
    float s = 0.f;
#pragma unroll
    for (int d = 0; d < DD; ++d) s = fmaf(r[d], r[d], s);
    norms[g] = s;
  } else {
    int i = g - KK;
    counts[i] = 0.f;
    if (i == 0) *loss = 0.f;
  }
}

// ---------------------------------------------------------------------------
// K2: pack codebook into WMMA B-fragment images (bf16), 4 KB per code tile.
// Element order within tile ct:
//   g_tile = ((s*2 + hh)*32 + lane)*8 + e8
//   value  = cb[code = ct*16 + (lane&15), d = s*32 + 16*(lane>>4) + hh*8 + e8]
// so in LDS each lane reads two conflict-free lane-stride-16B ds_load_b128s
// per s-step and concatenates them into the v16bf B fragment.
// ---------------------------------------------------------------------------
__global__ __launch_bounds__(256) void k_pack_cb(const float* __restrict__ cb,
                                                 __bf16* __restrict__ cbf) {
  int g0 = (blockIdx.x * 256 + threadIdx.x) * 4;
#pragma unroll
  for (int q = 0; q < 4; ++q) {
    int g    = g0 + q;
    int e8   = g & 7;
    int lane = (g >> 3) & 31;
    int sub  = (g >> 8) & 7;           // s*2 + hh
    int ct   = g >> 11;
    int s    = sub >> 1;
    int hh   = sub & 1;
    int code = ct * 16 + (lane & 15);
    int d    = s * 32 + 16 * (lane >> 4) + hh * 8 + e8;
    cbf[g] = (__bf16)cb[(size_t)code * DD + d];
  }
}

// ---------------------------------------------------------------------------
// K3: fused distance GEMM + argmin.  Block = 128 rows x all 2048 codes.
// 8 waves = 8 row-blocks of 16 rows; all waves march the same code tile,
// which is async-copied into a double-buffered LDS stage.
// ---------------------------------------------------------------------------
__global__ __launch_bounds__(256) void k_vq_argmin(const float* __restrict__ z,
                                                   const __bf16* __restrict__ cbf,
                                                   const float* __restrict__ norms,
                                                   int*   __restrict__ idx_g,
                                                   float* __restrict__ counts,
                                                   float* __restrict__ out_idx_f) {
  __shared__ float lds_z[DD * ROWS];      // raw f32 tile, [d][row], 64 KB
  __shared__ float lds_norms[KK];         // 8 KB
  __shared__ uint4 lds_bt[2][256];        // B tile double buffer, 2 x 4 KB

  const int tid = threadIdx.x;
  const int n0  = blockIdx.x * ROWS;
  const int b   = n0 / TT;
  const int t0  = n0 % TT;

  // ---- async stage: z tile (128 rows x 128 d, coalesced b128 over t) ----
  {
    const int tg = (tid & 31) * 4;        // t offset (0..124 step 4)
    const int dg = tid >> 5;              // 0..7
    const float* zb = z + (size_t)b * DD * TT + t0;
#pragma unroll
    for (int it = 0; it < 16; ++it) {
      int d = it * 8 + dg;
      async_ld_b128(zb + (size_t)d * TT + tg, &lds_z[d * ROWS + tg]);
    }
  }
  // ---- async stage: code norms ----
#pragma unroll
  for (int q = 0; q < 2; ++q) {
    int i = (q * 256 + tid) * 4;
    async_ld_b128(norms + i, &lds_norms[i]);
  }
  // ---- async stage: first B tile ----
  async_ld_b128((const char*)cbf + tid * 16, &lds_bt[0][tid]);
  wait_async_0();
  __syncthreads();

  const int lane = tid & 31;
  const int rb   = tid >> 5;              // row-block 0..7
  const int h    = lane >> 4;
  const int l15  = lane & 15;

  // Build A fragments once (lane holds M=lane&15, contiguous k order as in B;
  // element e -> k = e + 16*(lane>>4), per 32-wide k-step s).
  v16bf afrag[4];
#pragma unroll
  for (int s = 0; s < 4; ++s) {
#pragma unroll
    for (int e = 0; e < 16; ++e) {
      int k = e + 16 * h + s * 32;
      afrag[s][e] = (__bf16)lds_z[k * ROWS + rb * 16 + l15];
    }
  }

  float minv[8];
  int   mini[8];
#pragma unroll
  for (int j = 0; j < 8; ++j) { minv[j] = 3.4e38f; mini[j] = 0; }

  for (int ct = 0; ct < NTILE; ++ct) {
    // prefetch next B tile into the other buffer, then wait for tile ct
    if (ct + 1 < NTILE) {
      async_ld_b128((const char*)cbf + (size_t)(ct + 1) * 4096 + tid * 16,
                    &lds_bt[(ct + 1) & 1][tid]);
      wait_async_le1();
    } else {
      wait_async_0();
    }
    __syncthreads();                       // tile ct fully in LDS, all waves

    const char* bbase = (const char*)&lds_bt[ct & 1][0];
    v8f acc = {};
#pragma unroll
    for (int s = 0; s < 4; ++s) {
      v8bf lo = *(const v8bf*)(bbase + ((s * 2 + 0) * 32 + lane) * 16);
      v8bf hi = *(const v8bf*)(bbase + ((s * 2 + 1) * 32 + lane) * 16);
      v16bf bfrag = __builtin_shufflevector(lo, hi, 0, 1, 2, 3, 4, 5, 6, 7,
                                            8, 9, 10, 11, 12, 13, 14, 15);
      acc = __builtin_amdgcn_wmma_f32_16x16x32_bf16(
          /*neg_a=*/false, afrag[s], /*neg_b=*/false, bfrag,
          /*c_mod=*/(short)0, acc, /*reuse_a=*/false, /*reuse_b=*/false);
    }
    // C layout: VGPR j -> row m = rb*16 + 8*h + j, column code = ct*16 + l15
    int   code = ct * 16 + l15;
    float bias = lds_norms[code];
#pragma unroll
    for (int j = 0; j < 8; ++j) {
      float dst = __builtin_fmaf(-2.0f, acc[j], bias);
      if (dst < minv[j]) { minv[j] = dst; mini[j] = code; }
    }
    __syncthreads();                       // all waves done reading buf[ct&1]
  }

  // Butterfly min-reduce across the 16 lanes holding the same rows (wave32).
#pragma unroll
  for (int off = 1; off < 16; off <<= 1) {
#pragma unroll
    for (int j = 0; j < 8; ++j) {
      float ov = __shfl_xor(minv[j], off, 32);
      int   oi = __shfl_xor(mini[j], off, 32);
      if (ov < minv[j] || (ov == minv[j] && oi < mini[j])) {
        minv[j] = ov; mini[j] = oi;
      }
    }
  }
  if (l15 == 0) {
#pragma unroll
    for (int j = 0; j < 8; ++j) {
      int n = n0 + rb * 16 + 8 * h + j;
      idx_g[n]     = mini[j];
      out_idx_f[n] = (float)mini[j];
      atomicAdd(&counts[mini[j]], 1.0f);
    }
  }
}

// ---------------------------------------------------------------------------
// K4: z_q gather (coalesced writes, codebook L2-resident) + commitment loss.
// z_q_st value == z_q exactly (straight-through is identity in value).
// ---------------------------------------------------------------------------
__global__ __launch_bounds__(256) void k_gather_loss(const float* __restrict__ z,
                                                     const float* __restrict__ cb,
                                                     const int*   __restrict__ idx_g,
                                                     float* __restrict__ out_zq,
                                                     float* __restrict__ loss_acc) {
  const int tid = threadIdx.x;
  const int n0  = blockIdx.x * 64;
  const int b   = n0 / TT, t0 = n0 % TT;
  const int tl  = tid & 63, dg = tid >> 6;
  const int code = idx_g[n0 + tl];
  const float* zb  = z      + (size_t)b * DD * TT + t0 + tl;
  float*       ob  = out_zq + (size_t)b * DD * TT + t0 + tl;
  const float* cbr = cb + (size_t)code * DD;
  float acc = 0.f;
#pragma unroll
  for (int it = 0; it < 32; ++it) {
    int d = it * 4 + dg;
    float cv = cbr[d];
    float zv = zb[(size_t)d * TT];
    ob[(size_t)d * TT] = cv;
    float df = zv - cv;
    acc = fmaf(df, df, acc);
  }
#pragma unroll
  for (int off = 16; off >= 1; off >>= 1) acc += __shfl_xor(acc, off, 32);
  __shared__ float wsum[8];
  if ((tid & 31) == 0) wsum[tid >> 5] = acc;
  __syncthreads();
  if (tid == 0) {
    float s = 0.f;
#pragma unroll
    for (int i = 0; i < 8; ++i) s += wsum[i];
    atomicAdd(loss_acc, s);
  }
}

// ---------------------------------------------------------------------------
// K5: perplexity + loss scaling.
// ---------------------------------------------------------------------------
__global__ __launch_bounds__(256) void k_finalize(const float* __restrict__ counts,
                                                  const float* __restrict__ loss_acc,
                                                  float* __restrict__ out_loss,
                                                  float* __restrict__ out_perp) {
  const int tid = threadIdx.x;
  float hsum = 0.f;
#pragma unroll
  for (int i = 0; i < 8; ++i) {
    float p = counts[i * 256 + tid] * (1.0f / (float)NN);
    hsum += p * __logf(p + 1e-10f);
  }
#pragma unroll
  for (int off = 16; off >= 1; off >>= 1) hsum += __shfl_xor(hsum, off, 32);
  __shared__ float wsum[8];
  if ((tid & 31) == 0) wsum[tid >> 5] = hsum;
  __syncthreads();
  if (tid == 0) {
    float s = 0.f;
#pragma unroll
    for (int i = 0; i < 8; ++i) s += wsum[i];
    *out_perp = __expf(-s);
    *out_loss = loss_acc[0] * (0.25f / ((float)NN * (float)DD));
  }
}

// ---------------------------------------------------------------------------
extern "C" void kernel_launch(void* const* d_in, const int* in_sizes, int n_in,
                              void* d_out, int out_size, void* d_ws, size_t ws_size,
                              hipStream_t stream) {
  const float* z  = (const float*)d_in[0];   // (B, D, T) f32
  const float* cb = (const float*)d_in[1];   // (K, D)    f32

  float* out      = (float*)d_out;
  float* out_zq   = out;                               // B*D*T
  float* out_loss = out + (size_t)BB * DD * TT;        // 1
  float* out_perp = out_loss + 1;                      // 1
  float* out_idx  = out_perp + 1;                      // N (as float values)

  char*   ws     = (char*)d_ws;
  __bf16* cbf    = (__bf16*)(ws + WS_CBF);
  float*  norms  = (float*)(ws + WS_NORMS);
  float*  counts = (float*)(ws + WS_COUNTS);
  float*  loss   = (float*)(ws + WS_LOSS);
  int*    idx_g  = (int*)(ws + WS_IDX);

  k_prep       <<<16,        256, 0, stream>>>(cb, norms, counts, loss);
  k_pack_cb    <<<256,       256, 0, stream>>>(cb, cbf);
  k_vq_argmin  <<<NN / ROWS, 256, 0, stream>>>(z, cbf, norms, idx_g, counts, out_idx);
  k_gather_loss<<<NN / 64,   256, 0, stream>>>(z, cb, idx_g, out_zq, loss);
  k_finalize   <<<1,         256, 0, stream>>>(counts, loss, out_loss, out_perp);
}
